// Model_16801912062040
// MI455X (gfx1250) — matrix operand order repaired and checked by reading, not compiled, not run on hardware
//
#include <hip/hip_runtime.h>
#include <hip/hip_bf16.h>

// ---------------- static model config ----------------
#define B_    16
#define L_    192
#define CIN_  21
#define D_    128
#define P_    24
#define S_    12
#define NT_   16
#define PRED_ 96
#define COUT_ 21

#define G_    (B_ * D_)        // 2048 dynamic filters
#define HO_   14               // NT_ - 2
#define WO_   22               // 24 - 2
#define NPOS_ (HO_ * WO_)      // 308
#define N1_   (B_ * NPOS_)     // 4928  (GEMM1 N)
#define K1_   (D_ * 9)         // 1152  (GEMM1 K)
#define K2_   (G_ * 9)         // 18432 (GEMM2 K)
#define N2_   (B_ * NT_ * 24)  // 6144  (GEMM2 N)

// zero-padded y: (B, G, 18, 26) f16, pad=2 on both spatial dims
#define HP_   18
#define WP_   26
#define YPST_ (HP_ * WP_)      // 468 per (n,g) plane
#define YPAD_ELEMS ((size_t)B_ * G_ * YPST_)

typedef _Float16 half_t;
typedef _Float16 v16h __attribute__((ext_vector_type(16)));
typedef float    v8f  __attribute__((ext_vector_type(8)));

// ---------------- stage 1: token embedding + positional ----------------
__global__ void k_embed(const float* __restrict__ x, const float* __restrict__ tw,
                        float* __restrict__ embT) {
  int idx = blockIdx.x * blockDim.x + threadIdx.x;
  if (idx >= B_ * D_ * L_) return;
  int l  = idx % L_;
  int bd = idx / L_;
  int d  = bd % D_;
  int b  = bd / D_;
  float acc = 0.f;
#pragma unroll
  for (int k = 0; k < 3; ++k) {
    int lp = l + k - 1;
    lp = (lp + L_) % L_;                       // circular pad
    const float* xr = x + (b * L_ + lp) * CIN_;
    const float* wr = tw + d * CIN_ * 3 + k;   // token_w[d][cin][k]
    for (int c = 0; c < CIN_; ++c) acc += xr[c] * wr[c * 3];
  }
  int de = d & ~1;
  float freq = __expf(-(float)de * (9.210340371976184f / (float)D_)); // ln(1e4)
  float ang  = (float)l * freq;
  acc += (d & 1) ? __cosf(ang) : __sinf(ang);
  embT[(b * D_ + d) * L_ + l] = acc;
}

// ---------------- stage 2: patch embedding -> x5 (B, D, NT, 2, 12) ----------------
__global__ void k_patch(const float* __restrict__ embT, const float* __restrict__ pw,
                        const float* __restrict__ pb, float* __restrict__ x5) {
  int idx = blockIdx.x * blockDim.x + threadIdx.x;
  if (idx >= B_ * D_ * NT_ * 24) return;
  int b  = idx / (D_ * NT_ * 24);
  int i  = idx % (D_ * NT_ * 24);          // ((c*16+t)*2+j)*12+q
  int c  = i / 384;
  int r1 = i % 384;
  int t  = r1 / 24;
  int hw = r1 % 24;
  int j  = hw / 12, q = hw % 12;
  int pprime = 2 * q + j;                  // stack even/odd inverse
  int i2 = (t * P_ + pprime) * D_ + c;     // torch C-order reshape
  int d  = i2 / (NT_ * P_);
  int r2 = i2 % (NT_ * P_);
  int nt = r2 / P_;
  int p  = r2 % P_;
  const float* seq = embT + (b * D_ + d) * L_;
  float acc = pb[p];
  int base = nt * S_;
#pragma unroll
  for (int k = 0; k < P_; ++k) {
    int m = base + k;
    if (m > L_ - 1) m = L_ - 1;            // replicate pad
    acc += seq[m] * pw[p * P_ + k];
  }
  x5[idx] = acc;
}

// ---------------- per-layer: pool over (h,w) ----------------
__global__ void k_poolhw(const float* __restrict__ x5, float* __restrict__ pp) {
  int idx = blockIdx.x * blockDim.x + threadIdx.x;   // (b*128+c)*16+t
  if (idx >= B_ * D_ * NT_) return;
  const float* src = x5 + idx * 24;
  float s = 0.f;
#pragma unroll
  for (int k = 0; k < 24; ++k) s += src[k];
  pp[idx] = s * (1.f / 24.f);
}

// ---------------- per-layer: period conv + BN + ReLU + pool + alpha ----------------
__global__ __launch_bounds__(128) void k_alpha(
    const float* __restrict__ pp, const float* __restrict__ pcw,
    const float* __restrict__ pcb, const float* __restrict__ g,
    const float* __restrict__ beta, const float* __restrict__ aw,
    float* __restrict__ alpha) {
  __shared__ float ppL[D_ * NT_];
  __shared__ float pooled[D_];
  int b = blockIdx.x;
  int o = threadIdx.x;
  for (int i = o; i < D_ * NT_; i += D_) ppL[i] = pp[b * D_ * NT_ + i];
  __syncthreads();
  float acc[NT_];
#pragma unroll
  for (int t = 0; t < NT_; ++t) acc[t] = pcb[o];
  for (int c = 0; c < D_; ++c) {
    float w0 = pcw[(o * D_ + c) * 3 + 0];
    float w1 = pcw[(o * D_ + c) * 3 + 1];
    float w2 = pcw[(o * D_ + c) * 3 + 2];
    const float* pr = ppL + c * NT_;
#pragma unroll
    for (int t = 0; t < NT_; ++t) {   // scatter: pp[t] -> out[t-1..t+1] (zero pad)
      float v = pr[t];
      acc[t] += v * w1;
      if (t + 1 < NT_) acc[t + 1] += v * w0;
      if (t > 0)       acc[t - 1] += v * w2;
    }
  }
  float scale = g[o] * rsqrtf(1.f + 1e-5f);
  float bt = beta[o];
  float sum = 0.f;
#pragma unroll
  for (int t = 0; t < NT_; ++t) {
    float v = acc[t] * scale + bt;
    sum += fmaxf(v, 0.f);
  }
  pooled[o] = sum * (1.f / (float)NT_);
  __syncthreads();
  float a = 1.f;
  for (int c = 0; c < D_; ++c) a += aw[o * D_ + c] * pooled[c];
  alpha[b * D_ + o] = a;
}

// ---------------- per-layer: dynamic weights to f16 ----------------
__global__ void k_makeA1(const float* __restrict__ Wi_l, const float* __restrict__ alpha,
                         half_t* __restrict__ A1) {
  int idx = blockIdx.x * blockDim.x + threadIdx.x;
  if (idx >= G_ * K1_) return;
  int gg = idx / K1_, kk = idx % K1_;
  int c = kk / 9, rs = kk % 9;
  int b = gg >> 7, o = gg & 127;
  A1[idx] = (half_t)(Wi_l[(o * D_ + c) * 9 + rs] * alpha[b * D_ + o]);
}

__global__ void k_makeA2(const float* __restrict__ Wi_l, const float* __restrict__ alpha,
                         half_t* __restrict__ A2) {
  int idx = blockIdx.x * blockDim.x + threadIdx.x;
  if (idx >= D_ * K2_) return;
  int m = idx / K2_, kk = idx % K2_;
  int gg = kk / 9, rs = kk % 9;
  int r = rs / 3, s = rs % 3;
  int b = gg >> 7, o = gg & 127;
  A2[idx] = (half_t)(Wi_l[(o * D_ + m) * 9 + (2 - r) * 3 + (2 - s)] * alpha[b * D_ + o]);
}

// ---------------- zero fill (uint4 granularity) ----------------
__global__ void k_fill0(uint4* __restrict__ p, int n16) {
  int idx = blockIdx.x * blockDim.x + threadIdx.x;
  if (idx >= n16) return;
  uint4 z; z.x = 0; z.y = 0; z.z = 0; z.w = 0;
  p[idx] = z;
}

// ---------------- WMMA fragment helpers (ISA 16-bit layouts, wave32) ----------------
__device__ __forceinline__ v16h load_fragA(const half_t* As, int row0, int lane) {
  int row = row0 + (lane & 15);
  int hi  = lane >> 4;
  v16h a;
#pragma unroll
  for (int v = 0; v < 8; ++v) {
    int k = ((v >> 2) << 4) + hi * 8 + ((v & 3) << 1);
    a[2 * v]     = As[row * 32 + k];
    a[2 * v + 1] = As[row * 32 + k + 1];
  }
  return a;
}
__device__ __forceinline__ v16h load_fragB(const half_t* Bs, int col0, int lane) {
  int col = col0 + (lane & 15);
  int hi  = lane >> 4;
  v16h b;
#pragma unroll
  for (int v = 0; v < 8; ++v) {
    int k = ((v >> 2) << 4) + hi * 8 + ((v & 3) << 1);
    b[2 * v]     = Bs[col * 32 + k];
    b[2 * v + 1] = Bs[col * 32 + k + 1];
  }
  return b;
}

// ---------------- GEMM1: y = Wd * im2col(x5in) -> zero-padded f16 y ----------------
// M=2048, N=4928, K=1152; block tile 128x128, 8 waves (2x4), wave tile 64x32
// double-buffered LDS, one barrier per K-step, all gather geometry hoisted
__global__ __launch_bounds__(256) void k_gemm1(const half_t* __restrict__ A1,
                                               const float* __restrict__ x5in,
                                               half_t* __restrict__ ypad) {
  __shared__ __align__(16) half_t As[2][128 * 32];
  __shared__ __align__(16) half_t Bs[2][128 * 32];
  const int tid  = threadIdx.x;
  const int lane = tid & 31;
  const int wave = tid >> 5;
  const int wr = wave >> 2, wc = wave & 3;
  const int m0b = blockIdx.y * 128;
  const int n0b = blockIdx.x * 128;

  // ---- K-invariant gather geometry (hoisted) ----
  const int kk  = tid & 31;
  const int nl0 = tid >> 5;
  int  gbase[16];
  bool gvalid[16];
#pragma unroll
  for (int it = 0; it < 16; ++it) {
    int ncol = n0b + nl0 + it * 8;
    bool valid = ncol < N1_;
    int nc = valid ? ncol : 0;
    int n = nc / NPOS_, pos = nc % NPOS_;
    gbase[it]  = n * (D_ * NT_ * 24) + (pos / WO_) * 24 + (pos % WO_);
    gvalid[it] = valid;
  }
  // incremental decomposition of kg = k0 + kk into (c, rs): kg += 32 -> c += 3, rs += 5 mod 9
  int c_cur = kk / 9, rs_cur = kk % 9;

  const int arow = tid >> 1, aseg = tid & 1;
  const half_t* aptr = A1 + (m0b + arow) * K1_ + aseg * 16;

  const v8f vzero = {0.f, 0.f, 0.f, 0.f, 0.f, 0.f, 0.f, 0.f};
  v8f acc[4][2];
#pragma unroll
  for (int i = 0; i < 4; ++i)
#pragma unroll
    for (int j = 0; j < 2; ++j) acc[i][j] = vzero;

  // ---- prologue: stage tile 0 ----
  {
    const uint4* gp = reinterpret_cast<const uint4*>(aptr);
    uint4 u0 = gp[0], u1 = gp[1];
    uint4* sp = reinterpret_cast<uint4*>(&As[0][arow * 32 + aseg * 16]);
    sp[0] = u0; sp[1] = u1;
    int r0 = rs_cur / 3;
    int off = c_cur * 384 + r0 * 24 + (rs_cur - 3 * r0);
#pragma unroll
    for (int it = 0; it < 16; ++it) {
      float v = x5in[gbase[it] + off];
      Bs[0][(nl0 + it * 8) * 32 + kk] = (half_t)(gvalid[it] ? v : 0.f);
    }
  }

  const int KSTEPS = K1_ / 32;   // 36
  for (int ks = 0; ks < KSTEPS; ++ks) {
    __syncthreads();
    const int cur = ks & 1, nxt = cur ^ 1;
    // advance (c, rs) to next K-step
    int rs_n = rs_cur + 5;
    int c_n  = c_cur + 3;
    if (rs_n >= 9) { rs_n -= 9; c_n += 1; }
    // prefetch next tile into registers (overlaps with WMMA below)
    uint4 u0, u1;
    float gv[16];
    if (ks + 1 < KSTEPS) {
      const uint4* gp = reinterpret_cast<const uint4*>(aptr + (ks + 1) * 32);
      u0 = gp[0]; u1 = gp[1];
      int r0 = rs_n / 3;
      int off = c_n * 384 + r0 * 24 + (rs_n - 3 * r0);
#pragma unroll
      for (int it = 0; it < 16; ++it) gv[it] = x5in[gbase[it] + off];
    }
    // compute on current tile
    v16h af[4], bf[2];
#pragma unroll
    for (int i = 0; i < 4; ++i) af[i] = load_fragA(As[cur], wr * 64 + i * 16, lane);
#pragma unroll
    for (int j = 0; j < 2; ++j) bf[j] = load_fragB(Bs[cur], wc * 32 + j * 16, lane);
#pragma unroll
    for (int i = 0; i < 4; ++i)
#pragma unroll
      for (int j = 0; j < 2; ++j)
        acc[i][j] = __builtin_amdgcn_wmma_f32_16x16x32_f16(
            false, af[i], false, bf[j], (short)0, acc[i][j], false, false);
    // stage next tile
    if (ks + 1 < KSTEPS) {
      uint4* sp = reinterpret_cast<uint4*>(&As[nxt][arow * 32 + aseg * 16]);
      sp[0] = u0; sp[1] = u1;
#pragma unroll
      for (int it = 0; it < 16; ++it)
        Bs[nxt][(nl0 + it * 8) * 32 + kk] = (half_t)(gvalid[it] ? gv[it] : 0.f);
      c_cur = c_n; rs_cur = rs_n;
    }
  }

  // epilogue: f16 store into zero-padded y: ypad[n][m][hh+2][ww+2]
  int hi = lane >> 4, lcol = lane & 15;
#pragma unroll
  for (int j = 0; j < 2; ++j) {
    int ncol = n0b + wc * 32 + j * 16 + lcol;
    if (ncol >= N1_) continue;
    int n = ncol / NPOS_, pos = ncol % NPOS_;
    int pbase = n * (G_ * YPST_) + (pos / WO_ + 2) * WP_ + (pos % WO_ + 2);
#pragma unroll
    for (int i = 0; i < 4; ++i) {
#pragma unroll
      for (int e = 0; e < 8; ++e) {
        int m = m0b + wr * 64 + i * 16 + hi * 8 + e;
        ypad[pbase + m * YPST_] = (half_t)acc[i][j][e];
      }
    }
  }
}

// ---------------- GEMM2: z = Wt * im2col_pad2(y) + residual ----------------
// M=128, N=6144, K=18432; block tile 128x64, 8 waves (2x4), wave tile 64x16
// padded-y gathers are unconditional; double-buffered LDS
__global__ __launch_bounds__(256) void k_gemm2(const half_t* __restrict__ A2,
                                               const half_t* __restrict__ ypad,
                                               const float* __restrict__ x5in,
                                               float* __restrict__ x5out) {
  __shared__ __align__(16) half_t As[2][128 * 32];
  __shared__ __align__(16) half_t Bs[2][64 * 32];
  const int tid  = threadIdx.x;
  const int lane = tid & 31;
  const int wave = tid >> 5;
  const int wr = wave >> 2, wc = wave & 3;
  const int n0b = blockIdx.x * 64;

  // ---- K-invariant gather geometry (hoisted): ncol -> n,h,w ----
  const int kk  = tid & 31;
  const int nl0 = tid >> 5;
  int gbase[8];
#pragma unroll
  for (int it = 0; it < 8; ++it) {
    int ncol = n0b + nl0 + it * 8;
    int n = ncol / (NT_ * 24);
    int p = ncol % (NT_ * 24);
    int h = p / 24, w = p % 24;
    // padded addr = ((n*G + g)*HP + (h + r))*WP + (w + s); pad offset +2 cancels r-2
    gbase[it] = n * (G_ * YPST_) + h * WP_ + w;
  }
  int g_cur = kk / 9, rs_cur = kk % 9;

  const int arow = tid >> 1, aseg = tid & 1;
  const half_t* aptr = A2 + arow * K2_ + aseg * 16;

  const v8f vzero = {0.f, 0.f, 0.f, 0.f, 0.f, 0.f, 0.f, 0.f};
  v8f acc[4];
#pragma unroll
  for (int i = 0; i < 4; ++i) acc[i] = vzero;

  // ---- prologue: stage tile 0 ----
  {
    const uint4* gp = reinterpret_cast<const uint4*>(aptr);
    uint4 u0 = gp[0], u1 = gp[1];
    uint4* sp = reinterpret_cast<uint4*>(&As[0][arow * 32 + aseg * 16]);
    sp[0] = u0; sp[1] = u1;
    int r0 = rs_cur / 3;
    int off = g_cur * YPST_ + r0 * WP_ + (rs_cur - 3 * r0);
#pragma unroll
    for (int it = 0; it < 8; ++it)
      Bs[0][(nl0 + it * 8) * 32 + kk] = ypad[gbase[it] + off];
  }

  const int KSTEPS = K2_ / 32;   // 576
  for (int ks = 0; ks < KSTEPS; ++ks) {
    __syncthreads();
    const int cur = ks & 1, nxt = cur ^ 1;
    int rs_n = rs_cur + 5;
    int g_n  = g_cur + 3;
    if (rs_n >= 9) { rs_n -= 9; g_n += 1; }
    uint4 u0, u1;
    half_t gv[8];
    if (ks + 1 < KSTEPS) {
      const uint4* gp = reinterpret_cast<const uint4*>(aptr + (ks + 1) * 32);
      u0 = gp[0]; u1 = gp[1];
      int r0 = rs_n / 3;
      int off = g_n * YPST_ + r0 * WP_ + (rs_n - 3 * r0);
#pragma unroll
      for (int it = 0; it < 8; ++it) gv[it] = ypad[gbase[it] + off];
    }
    v16h af[4];
#pragma unroll
    for (int i = 0; i < 4; ++i) af[i] = load_fragA(As[cur], wr * 64 + i * 16, lane);
    v16h bf = load_fragB(Bs[cur], wc * 16, lane);
#pragma unroll
    for (int i = 0; i < 4; ++i)
      acc[i] = __builtin_amdgcn_wmma_f32_16x16x32_f16(
          false, af[i], false, bf, (short)0, acc[i], false, false);
    if (ks + 1 < KSTEPS) {
      uint4* sp = reinterpret_cast<uint4*>(&As[nxt][arow * 32 + aseg * 16]);
      sp[0] = u0; sp[1] = u1;
#pragma unroll
      for (int it = 0; it < 8; ++it)
        Bs[nxt][(nl0 + it * 8) * 32 + kk] = gv[it];
      g_cur = g_n; rs_cur = rs_n;
    }
  }

  // epilogue: add residual, write f32 x5out
  int hi = lane >> 4, lcol = lane & 15;
  int ncol = n0b + wc * 16 + lcol;
  int n = ncol / (NT_ * 24);
  int p = ncol % (NT_ * 24);
#pragma unroll
  for (int i = 0; i < 4; ++i) {
#pragma unroll
    for (int e = 0; e < 8; ++e) {
      int m = wr * 64 + i * 16 + hi * 8 + e;   // output channel 0..127
      int o = (n * D_ + m) * (NT_ * 24) + p;
      x5out[o] = acc[i][e] + x5in[o];
    }
  }
}

// ---------------- heads ----------------
__global__ void k_head1(const float* __restrict__ x5, const float* __restrict__ fc1w,
                        const float* __restrict__ fc1b, float* __restrict__ y1) {
  int idx = blockIdx.x * blockDim.x + threadIdx.x;   // (b*128+dd)*96+pr
  if (idx >= B_ * D_ * PRED_) return;
  int b  = idx / (D_ * PRED_);
  int r  = idx % (D_ * PRED_);
  int dd = r / PRED_;
  int pr = r % PRED_;
  float acc = fc1b[pr];
  const float* xb = x5 + b * (2 * L_ * D_);
  for (int m = 0; m < 2 * L_; ++m) acc += xb[m * D_ + dd] * fc1w[pr * (2 * L_) + m];
  y1[idx] = acc;
}

__global__ void k_head2(const float* __restrict__ y1, const float* __restrict__ fc2w,
                        const float* __restrict__ fc2b, float* __restrict__ out) {
  int idx = blockIdx.x * blockDim.x + threadIdx.x;   // (b*96+pr)*21+co
  if (idx >= B_ * PRED_ * COUT_) return;
  int b  = idx / (PRED_ * COUT_);
  int r  = idx % (PRED_ * COUT_);
  int pr = r / COUT_;
  int co = r % COUT_;
  float acc = fc2b[co];
  for (int dd = 0; dd < D_; ++dd)
    acc += y1[(b * D_ + dd) * PRED_ + pr] * fc2w[co * D_ + dd];
  out[idx] = acc;
}

// ---------------- launch ----------------
extern "C" void kernel_launch(void* const* d_in, const int* in_sizes, int n_in,
                              void* d_out, int out_size, void* d_ws, size_t ws_size,
                              hipStream_t stream) {
  (void)in_sizes; (void)n_in; (void)out_size; (void)ws_size;
  const float* x       = (const float*)d_in[0];
  const float* token_w = (const float*)d_in[1];
  const float* patch_w = (const float*)d_in[2];
  const float* patch_b = (const float*)d_in[3];
  const float* Wi      = (const float*)d_in[4];
  const float* pconv_w = (const float*)d_in[5];
  const float* pconv_b = (const float*)d_in[6];
  const float* bn_g    = (const float*)d_in[7];
  const float* bn_b    = (const float*)d_in[8];
  const float* aconv_w = (const float*)d_in[9];
  const float* fc1_w   = (const float*)d_in[10];
  const float* fc1_b   = (const float*)d_in[11];
  const float* fc2_w   = (const float*)d_in[12];
  const float* fc2_b   = (const float*)d_in[13];

  char* ws = (char*)d_ws;
  size_t off = 0;
  float*  emb   = (float*)(ws + off); off += (size_t)B_ * D_ * L_ * 4;          // 1.5 MB
  float*  x5a   = (float*)(ws + off); off += (size_t)B_ * D_ * NT_ * 24 * 4;    // 3 MB
  float*  x5b   = (float*)(ws + off); off += (size_t)B_ * D_ * NT_ * 24 * 4;    // 3 MB
  float*  pp    = (float*)(ws + off); off += (size_t)B_ * D_ * NT_ * 4;
  float*  alpha = (float*)(ws + off); off += (size_t)B_ * D_ * 4;
  half_t* A1    = (half_t*)(ws + off); off += (size_t)G_ * K1_ * 2;             // 4.7 MB
  half_t* A2    = (half_t*)(ws + off); off += (size_t)D_ * K2_ * 2;             // 4.7 MB
  half_t* ypad  = (half_t*)(ws + off); off += YPAD_ELEMS * 2;                   // 30.7 MB
  float*  y1    = (float*)(ws + off); off += (size_t)B_ * D_ * PRED_ * 4;

  const int TB = 256;
  k_embed<<<(B_ * D_ * L_ + TB - 1) / TB, TB, 0, stream>>>(x, token_w, emb);
  k_patch<<<(B_ * D_ * NT_ * 24 + TB - 1) / TB, TB, 0, stream>>>(emb, patch_w, patch_b, x5a);

  // zero the padded-y buffer once; GEMM1 only ever writes the interior
  int n16 = (int)(YPAD_ELEMS * 2 / 16);
  k_fill0<<<(n16 + TB - 1) / TB, TB, 0, stream>>>((uint4*)ypad, n16);

  for (int layer = 0; layer < 2; ++layer) {
    const float* xin  = (layer == 0) ? x5a : x5b;
    float*       xout = (layer == 0) ? x5b : x5a;
    const float* Wi_l  = Wi      + (size_t)layer * D_ * D_ * 9;
    const float* pcw_l = pconv_w + (size_t)layer * D_ * D_ * 3;
    const float* pcb_l = pconv_b + (size_t)layer * D_;
    const float* g_l   = bn_g    + (size_t)layer * D_;
    const float* bt_l  = bn_b    + (size_t)layer * D_;
    const float* aw_l  = aconv_w + (size_t)layer * D_ * D_;

    k_poolhw<<<(B_ * D_ * NT_ + TB - 1) / TB, TB, 0, stream>>>(xin, pp);
    k_alpha<<<B_, D_, 0, stream>>>(pp, pcw_l, pcb_l, g_l, bt_l, aw_l, alpha);
    k_makeA1<<<(G_ * K1_ + TB - 1) / TB, TB, 0, stream>>>(Wi_l, alpha, A1);
    dim3 g1((N1_ + 127) / 128, G_ / 128);
    k_gemm1<<<g1, TB, 0, stream>>>(A1, xin, ypad);
    k_makeA2<<<(D_ * K2_ + TB - 1) / TB, TB, 0, stream>>>(Wi_l, alpha, A2);
    k_gemm2<<<N2_ / 64, TB, 0, stream>>>(A2, ypad, xin, xout);
  }

  // after 2 layers the result is back in x5a
  k_head1<<<(B_ * D_ * PRED_ + TB - 1) / TB, TB, 0, stream>>>(x5a, fc1_w, fc1_b, y1);
  k_head2<<<(B_ * PRED_ * COUT_ + TB - 1) / TB, TB, 0, stream>>>(y1, fc2_w, fc2_b, (float*)d_out);
}
